// EMAVectorQuantizer_71047349010730
// MI455X (gfx1250) — compile-verified, matrix operand order checked by reading, not run on hardware
//
#include <hip/hip_runtime.h>
#include <hip/hip_bf16.h>

// Problem constants (match reference)
#define N_ROWS 16384        // 8 * 2048
#define DIM    256
#define KCODES 8192
#define DECAY  0.99f
#define EPSV   1e-05f
#define CW     0.25f

constexpr int TILE_COLS       = 16;
constexpr int PANELS          = 4;                               // A panels per wave
constexpr int ROWS_PER_WAVE   = 16 * PANELS;                     // 64
constexpr int WAVES_PER_WG    = 8;
constexpr int ROWS_PER_WG     = ROWS_PER_WAVE * WAVES_PER_WG;    // 512
constexpr int KSPLIT          = 8;
constexpr int TILES_PER_SPLIT = (KCODES / KSPLIT) / TILE_COLS;   // 64
constexpr int PAIRS_PER_COL   = DIM / 2;                         // 128 packed bf16 pairs
constexpr int TILE_U32        = TILE_COLS * PAIRS_PER_COL;       // 2048 u32 per packed tile
constexpr int LDS_PITCH       = 132;                             // pad: 16B-aligned, bank spread
constexpr int LDS_TILE        = TILE_COLS * LDS_PITCH;           // u32 per LDS buffer

typedef __bf16 v16bf __attribute__((ext_vector_type(16)));
typedef float  v8f   __attribute__((ext_vector_type(8)));
typedef int    v4i   __attribute__((vector_size(4 * sizeof(int))));

union BF16x16 { v16bf v; unsigned u[8]; };

__device__ inline unsigned pack_bf16x2(float lo, float hi) {
    union { __bf16 b; unsigned short u; } cl, ch;
    cl.b = (__bf16)lo;
    ch.b = (__bf16)hi;
    return (unsigned)cl.u | ((unsigned)ch.u << 16);
}

// ---- CDNA5 async global->LDS copy (ASYNCcnt-tracked LDS DMA) -------------
__device__ inline void async_copy_b128(const unsigned* gsrc, unsigned* ldst) {
#if __has_builtin(__builtin_amdgcn_global_load_async_to_lds_b128)
    __builtin_amdgcn_global_load_async_to_lds_b128(
        (__attribute__((address_space(1))) v4i*)(size_t)gsrc,
        (__attribute__((address_space(3))) v4i*)(unsigned)(size_t)ldst,
        0, 0);
#else
    unsigned      lo = (unsigned)(size_t)ldst;          // LDS byte offset
    unsigned long long ga = (unsigned long long)(size_t)gsrc;
    asm volatile("global_load_async_to_lds_b128 %0, %1, off"
                 :: "v"(lo), "v"(ga) : "memory");
#endif
}

__device__ inline void wait_async_all() {
#if __has_builtin(__builtin_amdgcn_s_wait_asynccnt)
    __builtin_amdgcn_s_wait_asynccnt(0);
#else
    asm volatile("s_wait_asynccnt 0x0" ::: "memory");
#endif
}

// ---------------------------------------------------------------------------
// Kernel 1: zero the atomic accumulator region of the workspace
// ---------------------------------------------------------------------------
__global__ void vq_zero(float* __restrict__ p, int n) {
    int i = blockIdx.x * 256 + threadIdx.x;
    if (i < n) p[i] = 0.0f;
}

// ---------------------------------------------------------------------------
// Kernel 2: pre-pack the codebook to bf16 in tile-contiguous layout.
//   Tile t (16 columns): packed[t*2048 + col*128 + kp] = bf16x2(e[2kp], e[2kp+1])
// ---------------------------------------------------------------------------
__global__ __launch_bounds__(256)
void vq_pack(const float* __restrict__ emb, unsigned* __restrict__ packed) {
    const int t = blockIdx.x;                       // KCODES/16 tiles
    #pragma unroll
    for (int j = 0; j < 8; ++j) {
        int i   = threadIdx.x + j * 256;            // 0..2047
        int col = i >> 7;
        int kp  = i & 127;
        const float* er = emb + (size_t)(t * TILE_COLS + col) * DIM + kp * 2;
        packed[(size_t)t * TILE_U32 + col * PAIRS_PER_COL + kp] = pack_bf16x2(er[0], er[1]);
    }
}

// ---------------------------------------------------------------------------
// Kernel 3: squared L2 norm of each codebook row (fp32 exact)
// ---------------------------------------------------------------------------
__global__ __launch_bounds__(256)
void vq_esq(const float* __restrict__ emb, float* __restrict__ esq) {
    const int k = blockIdx.x;
    const int d = threadIdx.x;
    float v = emb[k * DIM + d];
    float s = v * v;
    #pragma unroll
    for (int m = 16; m >= 1; m >>= 1) s += __shfl_xor(s, m, 32);
    __shared__ float red[WAVES_PER_WG];
    if ((d & 31) == 0) red[d >> 5] = s;
    __syncthreads();
    if (d == 0) {
        float t = 0.0f;
        #pragma unroll
        for (int w = 0; w < WAVES_PER_WG; ++w) t += red[w];
        esq[k] = t;
    }
}

// ---------------------------------------------------------------------------
// Kernel 4: fused bf16-WMMA distance GEMM + running argmin.
//   grid = (N_ROWS/512, KSPLIT), block = 256 (8 waves).
//   FOUR 16-row A panels per wave (registers, WMMA layout) -> each B fragment
//   feeds 4 independent WMMA chains (0.5 ds_load_b128 per WMMA). Codebook
//   tiles streamed into double-buffered LDS with async LDS-DMA (ASYNCcnt);
//   B fragments double-buffered in registers one chunk ahead.
//   dist = ||e||^2 - 2 * z.e   (||z||^2 constant per row, dropped for argmin)
// ---------------------------------------------------------------------------
__global__ __launch_bounds__(256)
void vq_argmin_wmma(const float* __restrict__ z,
                    const unsigned* __restrict__ packed,
                    const float* __restrict__ esq,
                    float* __restrict__ partDist,
                    int*   __restrict__ partIdx)
{
    __shared__ unsigned ldsB[2 * LDS_TILE];   // double-buffered packed bf16 pairs

    const int tid  = threadIdx.x;
    const int lane = tid & 31;
    const int wave = tid >> 5;
    const int half = lane >> 4;    // 0 or 1
    const int l16  = lane & 15;
    const int rowBase   = blockIdx.x * ROWS_PER_WG + wave * ROWS_PER_WAVE;
    const int ysplit    = blockIdx.y;
    const int colStart  = ysplit * (KCODES / KSPLIT);
    const int tileStart = colStart / TILE_COLS;

    // Per-thread async share of a tile: 2 B128 transfers (512 total per tile).
    // e = tid + 256*j -> col = e>>5, qp = e&31 (B128 group within the column).
    const int sCol0 = tid >> 5,          sQp0 = tid & 31;
    const int sCol1 = (tid + 256) >> 5,  sQp1 = tid & 31;

    auto stage_tile = [&](int tileG, unsigned bufOff) {
        const unsigned* gt = packed + (size_t)tileG * TILE_U32;
        async_copy_b128(gt + sCol0 * PAIRS_PER_COL + sQp0 * 4,
                        &ldsB[bufOff + sCol0 * LDS_PITCH + sQp0 * 4]);
        async_copy_b128(gt + sCol1 * PAIRS_PER_COL + sQp1 * 4,
                        &ldsB[bufOff + sCol1 * LDS_PITCH + sQp1 * 4]);
    };

    // B fragment fetch: chunk c covers pairs [c*16, c*16+16) of this lane's
    // column; VGPR v reads pair c*16 + (v>>2)*8 + half*4 + (v&3).
    const unsigned bBase = l16 * LDS_PITCH + half * 4;
    auto load_b = [&](BF16x16& b, unsigned bufOff, int c) {
        #pragma unroll
        for (int v = 0; v < 8; ++v) {
            int kp = c * 16 + ((v >> 2) << 3) + (v & 3);
            b.u[v] = ldsB[bufOff + bBase + kp];
        }
    };

    // ---- Preload four A panels (64 rows of z, D=256) as bf16 in the exact
    //      16-bit A-matrix WMMA layout: chunk c covers K=32c..32c+31; within a
    //      chunk, VGPR v holds the pair K = (v>>2)*16 + half*8 + (v&3)*2.
    BF16x16 a[PANELS][8];
    #pragma unroll
    for (int p = 0; p < PANELS; ++p) {
        const float* zrow = z + (size_t)(rowBase + p * 16 + l16) * DIM;
        #pragma unroll
        for (int c = 0; c < 8; ++c) {
            #pragma unroll
            for (int v = 0; v < 8; ++v) {
                int k = c * 32 + ((v >> 2) << 4) + half * 8 + ((v & 3) << 1);
                a[p][c].u[v] = pack_bf16x2(zrow[k], zrow[k + 1]);
            }
        }
    }

    float bestD[PANELS][8];
    int   bestI[PANELS][8];
    #pragma unroll
    for (int p = 0; p < PANELS; ++p)
        #pragma unroll
        for (int r = 0; r < 8; ++r) { bestD[p][r] = 3.4e38f; bestI[p][r] = 0; }

    // Prologue: async-stage tile 0 into buffer 0.
    stage_tile(tileStart, 0);
    wait_async_all();
    __syncthreads();

    for (int t = 0; t < TILES_PER_SPLIT; ++t) {
        const int      colBase = colStart + t * TILE_COLS;
        const unsigned cur     = (unsigned)(t & 1) * LDS_TILE;
        const unsigned nxt     = cur ^ LDS_TILE;
        const bool     hasNext = (t + 1) < TILES_PER_SPLIT;

        // ---- kick off next tile's LDS-DMA; runs underneath the WMMAs
        if (hasNext) {
            stage_tile(tileStart + t + 1, nxt);
            if (t + 2 < TILES_PER_SPLIT) {
                __builtin_prefetch(packed + (size_t)(tileStart + t + 2) * TILE_U32 + tid * 2, 0, 1);
            }
        }

        // ---- 32 WMMAs: four independent accumulator chains sharing each B;
        //      B fragments double-buffered one chunk ahead in registers.
        v8f acc[PANELS];
        #pragma unroll
        for (int p = 0; p < PANELS; ++p) acc[p] = (v8f){};

        BF16x16 b[2];
        load_b(b[0], cur, 0);
        #pragma unroll
        for (int c = 0; c < 8; ++c) {
            if (c + 1 < 8) load_b(b[(c + 1) & 1], cur, c + 1);
            #pragma unroll
            for (int p = 0; p < PANELS; ++p) {
                acc[p] = __builtin_amdgcn_wmma_f32_16x16x32_bf16(
                             false, a[p][c].v, false, b[c & 1].v, (short)0,
                             acc[p], false, false);
            }
        }

        // ---- epilogue: distance + running argmin.
        //      C layout: lane column = l16; VGPR r -> row (half*8 + r).
        const int   colG = colBase + l16;
        const float eq   = esq[colG];
        #pragma unroll
        for (int p = 0; p < PANELS; ++p) {
            #pragma unroll
            for (int r = 0; r < 8; ++r) {
                float d = eq - 2.0f * acc[p][r];
                if (d < bestD[p][r]) { bestD[p][r] = d; bestI[p][r] = colG; }
            }
        }

        // ---- next tile's DMA must land before anyone reads it
        wait_async_all();
        __syncthreads();
    }

    // ---- min-reduce the 16 lanes sharing each row (lanes 0-15 and 16-31)
    #pragma unroll
    for (int p = 0; p < PANELS; ++p) {
        #pragma unroll
        for (int r = 0; r < 8; ++r) {
            float d = bestD[p][r];
            int   i = bestI[p][r];
            #pragma unroll
            for (int m = 8; m >= 1; m >>= 1) {
                float od = __shfl_xor(d, m, 32);
                int   oi = __shfl_xor(i, m, 32);
                if (od < d || (od == d && oi < i)) { d = od; i = oi; }
            }
            if (l16 == 0) {
                int row = rowBase + p * 16 + half * 8 + r;
                partDist[ysplit * N_ROWS + row] = d;
                partIdx [ysplit * N_ROWS + row] = i;
            }
        }
    }
}

// ---------------------------------------------------------------------------
// Kernel 5: combine the K-split partial winners
// ---------------------------------------------------------------------------
__global__ void vq_pick(const float* __restrict__ pd, const int* __restrict__ pi,
                        int* __restrict__ idxOut, float* __restrict__ idxFloatOut) {
    int i = blockIdx.x * 256 + threadIdx.x;
    if (i >= N_ROWS) return;
    float bd = pd[i];
    int   bi = pi[i];
    #pragma unroll
    for (int s = 1; s < KSPLIT; ++s) {
        float d = pd[s * N_ROWS + i];
        int   x = pi[s * N_ROWS + i];
        if (d < bd || (d == bd && x < bi)) { bd = d; bi = x; }
    }
    idxOut[i]      = bi;
    idxFloatOut[i] = (float)bi;
}

// ---------------------------------------------------------------------------
// Kernel 6: gather z_q, commitment-loss partial, segment sums (atomics)
//   grid = N_ROWS blocks, block = DIM threads.
// ---------------------------------------------------------------------------
__global__ __launch_bounds__(256)
void vq_scatter(const float* __restrict__ z, const float* __restrict__ emb,
                const int* __restrict__ idx,
                float* __restrict__ zq, float* __restrict__ cntAcc,
                float* __restrict__ embAcc, float* __restrict__ lossAcc)
{
    const int row = blockIdx.x;
    const int d   = threadIdx.x;
    const int k   = idx[row];
    const float e  = emb[(size_t)k * DIM + d];
    const float zv = z[(size_t)row * DIM + d];
    zq[(size_t)row * DIM + d] = e;   // straight-through output == gathered code

    float diff = zv - e;
    float s = diff * diff;
    #pragma unroll
    for (int m = 16; m >= 1; m >>= 1) s += __shfl_xor(s, m, 32);
    __shared__ float red[WAVES_PER_WG];
    if ((d & 31) == 0) red[d >> 5] = s;
    __syncthreads();
    if (d == 0) {
        float t = 0.0f;
        #pragma unroll
        for (int w = 0; w < WAVES_PER_WG; ++w) t += red[w];
        atomicAdd(lossAcc, t);
        atomicAdd(&cntAcc[k], 1.0f);
    }
    atomicAdd(&embAcc[(size_t)k * DIM + d], zv);
}

// ---------------------------------------------------------------------------
// Kernel 7: EMA updates + normalized embeddings + loss scalars
//   grid = KCODES blocks, block = DIM threads.
// ---------------------------------------------------------------------------
__global__ __launch_bounds__(256)
void vq_ema(const float* __restrict__ emaCS, const float* __restrict__ emaES,
            const float* __restrict__ cntAcc, const float* __restrict__ embAcc,
            const float* __restrict__ lossAcc,
            float* __restrict__ newEmb, float* __restrict__ newCS,
            float* __restrict__ newES, float* __restrict__ lossOut)
{
    const int k = blockIdx.x;
    const int d = threadIdx.x;
    const float ncs = DECAY * emaCS[k] + (1.0f - DECAY) * cntAcc[k];
    const float nes = DECAY * emaES[(size_t)k * DIM + d]
                    + (1.0f - DECAY) * embAcc[(size_t)k * DIM + d];
    newES [(size_t)k * DIM + d] = nes;
    newEmb[(size_t)k * DIM + d] = nes / (ncs + EPSV);
    if (d == 0) newCS[k] = ncs;
    if (k == 0 && d == 0) {
        float l = CW * lossAcc[0] / (float)((size_t)N_ROWS * DIM);
        lossOut[0] = l;
        lossOut[1] = l;
    }
}

// ---------------------------------------------------------------------------
// Host-side launcher
// ---------------------------------------------------------------------------
extern "C" void kernel_launch(void* const* d_in, const int* in_sizes, int n_in,
                              void* d_out, int out_size, void* d_ws, size_t ws_size,
                              hipStream_t stream) {
    (void)n_in; (void)out_size; (void)ws_size; (void)in_sizes;

    const float* z      = (const float*)d_in[0];   // (N, D)
    const float* emb    = (const float*)d_in[1];   // (K, D)
    const float* emaCS  = (const float*)d_in[2];   // (K,)
    const float* emaES  = (const float*)d_in[3];   // (K, D)

    // ---- output layout (floats, reference return order) ----
    float* out       = (float*)d_out;
    float* o_zq      = out;                                       // N*D
    float* o_idx     = o_zq + (size_t)N_ROWS * DIM;               // N
    float* o_loss    = o_idx + N_ROWS;                            // 2
    float* o_newEmb  = o_loss + 2;                                // K*D
    float* o_newCS   = o_newEmb + (size_t)KCODES * DIM;           // K
    float* o_newES   = o_newCS + KCODES;                          // K*D

    // ---- workspace layout ----
    // packed bf16 codebook first (keeps it 16B-aligned for B128 DMA)
    unsigned* w_packed = (unsigned*)d_ws;                         // K*D/2 u32 (4MB)
    float* ws        = (float*)(w_packed + (size_t)KCODES * PAIRS_PER_COL);
    float* w_esq     = ws;                                        // K
    float* w_pDist   = w_esq + KCODES;                            // KSPLIT*N
    int*   w_pIdx    = (int*)(w_pDist + KSPLIT * N_ROWS);         // KSPLIT*N
    int*   w_idx     = (int*)(w_pIdx + KSPLIT * N_ROWS);          // N
    float* w_cnt     = (float*)(w_idx + N_ROWS);                  // K
    float* w_embAcc  = w_cnt + KCODES;                            // K*D
    float* w_loss    = w_embAcc + (size_t)KCODES * DIM;           // 1

    // 1) zero atomic accumulators (cnt + embAcc + loss, contiguous)
    const int nZero = KCODES + KCODES * DIM + 1;
    vq_zero<<<(nZero + 255) / 256, 256, 0, stream>>>(w_cnt, nZero);

    // 2) pre-pack codebook to bf16 tiles (one-time, reused by all WGs/splits)
    vq_pack<<<KCODES / TILE_COLS, 256, 0, stream>>>(emb, w_packed);

    // 3) codebook squared norms
    vq_esq<<<KCODES, DIM, 0, stream>>>(emb, w_esq);

    // 4) fused bf16 WMMA distance GEMM + argmin (8-way K split)
    dim3 g4(N_ROWS / ROWS_PER_WG, KSPLIT);
    vq_argmin_wmma<<<g4, WAVES_PER_WG * 32, 0, stream>>>(z, w_packed, w_esq, w_pDist, w_pIdx);

    // 5) combine K-split winners
    vq_pick<<<(N_ROWS + 255) / 256, 256, 0, stream>>>(w_pDist, w_pIdx, w_idx, o_idx);

    // 6) gather z_q + loss + segment sums
    vq_scatter<<<N_ROWS, DIM, 0, stream>>>(z, emb, w_idx, o_zq, w_cnt, w_embAcc, w_loss);

    // 7) EMA updates + outputs
    vq_ema<<<KCODES, DIM, 0, stream>>>(emaCS, emaES, w_cnt, w_embAcc, w_loss,
                                       o_newEmb, o_newCS, o_newES, o_loss);
}
